// MultiHeadAttention_89318139888179
// MI455X (gfx1250) — matrix operand order repaired and checked by reading, not compiled, not run on hardware
//
#include <hip/hip_runtime.h>
#include <hip/hip_bf16.h>

typedef __bf16 bf16_t;
typedef __bf16 v4bf  __attribute__((ext_vector_type(4)));
typedef __bf16 v8bf  __attribute__((ext_vector_type(8)));
typedef __bf16 v16bf __attribute__((ext_vector_type(16)));
typedef float  v8f   __attribute__((ext_vector_type(8)));

#define B_   4
#define S_   2048
#define D_   1024
#define H_   16
#define HD_  64
#define BS_  (B_*S_)

// ---------------- WMMA helpers (CDNA5 wave32, 16x16x32 bf16 -> f32) ----------------

__device__ __forceinline__ v8f wmma_bf16(v16bf a, v16bf b, v8f c) {
    return __builtin_amdgcn_wmma_f32_16x16x32_bf16(
        /*neg_a=*/false, a, /*neg_b=*/false, b,
        /*c_mod=*/(short)0, c, /*reuse_a=*/false, /*reuse_b=*/false);
}

// A-matrix 16x32 (MxK): lane&15 = row M; lane>>4 = kg.
// Per-lane elements: K = kg*16 + {0..15} (contiguous, matches B fragments)
__device__ __forceinline__ v16bf load_a_frag(const bf16_t* __restrict__ X, int ld,
                                             int row0, int k0, int lane) {
    const int r  = row0 + (lane & 15);
    const int kg = lane >> 4;
    const bf16_t* p = X + (size_t)r * ld + k0 + kg * 16;
    v8bf lo = *(const v8bf*)p;
    v8bf hi = *(const v8bf*)(p + 8);
    v16bf out;
#pragma unroll
    for (int i = 0; i < 8; ++i) { out[i] = lo[i]; out[i + 8] = hi[i]; }
    return out;
}

// B-matrix 32x16 (KxN) where B[k][n] = Wrow[n][k] (Wrow row-major, row stride ld).
// lane&15 = col N; lane>>4 = kh; element i holds K = kh*16 + i (contiguous in row).
__device__ __forceinline__ v16bf load_b_frag(const bf16_t* __restrict__ Wrow, int ld,
                                             int n0, int k0, int lane) {
    const bf16_t* p = Wrow + (size_t)(n0 + (lane & 15)) * ld + k0 + (lane >> 4) * 16;
    return *(const v16bf*)p;  // 32B contiguous
}

// Fragment loads from LDS (rows must be 16B aligned)
__device__ __forceinline__ v16bf load_a_frag_lds(const bf16_t* P, int ld, int k0, int lane) {
    const bf16_t* p = P + (lane & 15) * ld + k0 + (lane >> 4) * 16;
    v8bf lo = *(const v8bf*)p;
    v8bf hi = *(const v8bf*)(p + 8);
    v16bf out;
#pragma unroll
    for (int i = 0; i < 8; ++i) { out[i] = lo[i]; out[i + 8] = hi[i]; }
    return out;
}

__device__ __forceinline__ v16bf load_b_frag_lds(const bf16_t* T, int ld, int n0,
                                                 int k0, int lane) {
    const bf16_t* p = T + (size_t)(n0 + (lane & 15)) * ld + k0 + (lane >> 4) * 16;
    v8bf x0 = *(const v8bf*)p;
    v8bf x1 = *(const v8bf*)(p + 8);
    v16bf out;
#pragma unroll
    for (int i = 0; i < 8; ++i) { out[i] = x0[i]; out[i + 8] = x1[i]; }
    return out;
}

// Async global->LDS copy (CDNA5): LDS destination address comes from a VGPR; generic LDS
// pointer's low 32 bits are the in-aperture LDS address (ISA 10.2).
__device__ __forceinline__ void async_copy_b128(const bf16_t* gptr, bf16_t* lptr) {
    unsigned lds = (unsigned)(unsigned long long)lptr;
    asm volatile("global_load_async_to_lds_b128 %0, %1, off"
                 :: "v"(lds), "v"(gptr) : "memory");
}
__device__ __forceinline__ void wait_asynccnt0() {
    asm volatile("s_wait_asynccnt 0x0" ::: "memory");
}

// lane-xor shuffle via ds_swizzle (group-of-32 mode: and=0x1f, or=0, xor=m)
#define SWZ_XOR_F(x, m) \
    __int_as_float(__builtin_amdgcn_ds_swizzle(__float_as_int(x), (0x1f | ((m) << 10))))

// ---------------- fp32 -> bf16 conversion ----------------

__global__ __launch_bounds__(256) void cvt_f32_bf16(const float* __restrict__ src,
                                                    bf16_t* __restrict__ dst, int n) {
    int i = (blockIdx.x * 256 + threadIdx.x) * 4;
    if (i < n) {
        float4 v = *(const float4*)(src + i);
        v4bf o;
        o[0] = (bf16_t)v.x; o[1] = (bf16_t)v.y; o[2] = (bf16_t)v.z; o[3] = (bf16_t)v.w;
        *(v4bf*)(dst + i) = o;
    }
}

// ---------------- GEMM: Y[m][e] = sum_k X[m][k]*W[e][k] + bias[e] ----------------
// mode 0: bf16 output scattered to (B,H,S,HD);  mode 1: f32 output to (BS,D) flat.
// Grid: (BS/128, D/128), 256 threads = 8 waves; wave tile 32(M) x 64(N).

__global__ __launch_bounds__(256) void gemm_bf16(const bf16_t* __restrict__ X,
                                                 const bf16_t* __restrict__ W,
                                                 const float* __restrict__ bias,
                                                 bf16_t* __restrict__ out_bhsd,
                                                 float* __restrict__ out_f32,
                                                 int mode) {
    const int lane = threadIdx.x & 31;
    const int wave = threadIdx.x >> 5;
    const int row0 = blockIdx.x * 128 + (wave & 3) * 32;
    const int col0 = blockIdx.y * 128 + (wave >> 2) * 64;

    v8f acc[2][4];
#pragma unroll
    for (int i = 0; i < 2; ++i)
#pragma unroll
        for (int j = 0; j < 4; ++j) acc[i][j] = (v8f)(0.0f);

    for (int k0 = 0; k0 < D_; k0 += 32) {
        if (k0 + 256 < D_) {  // WGP-scope streaming prefetch (global_prefetch_b8)
            __builtin_prefetch(X + (size_t)(row0 + lane) * D_ + k0 + 256, 0, 3);
            __builtin_prefetch(W + (size_t)(col0 + lane * 2) * D_ + k0 + 256, 0, 3);
        }
        v16bf a0 = load_a_frag(X, D_, row0,      k0, lane);
        v16bf a1 = load_a_frag(X, D_, row0 + 16, k0, lane);
#pragma unroll
        for (int j = 0; j < 4; ++j) {
            v16bf b = load_b_frag(W, D_, col0 + 16 * j, k0, lane);
            acc[0][j] = wmma_bf16(a0, b, acc[0][j]);
            acc[1][j] = wmma_bf16(a1, b, acc[1][j]);
        }
    }

    const int rhalf = lane >> 4, cl = lane & 15;
#pragma unroll
    for (int i = 0; i < 2; ++i) {
#pragma unroll
        for (int j = 0; j < 4; ++j) {
            const int e = col0 + j * 16 + cl;
            const float bv = bias[e];
#pragma unroll
            for (int r = 0; r < 8; ++r) {
                const int row = row0 + i * 16 + rhalf * 8 + r;
                const float v = acc[i][j][r] + bv;
                if (mode == 0) {
                    const int b = row >> 11, s = row & (S_ - 1);
                    const int h = e >> 6,  hd = e & 63;
                    out_bhsd[(((size_t)(b * H_ + h) * S_ + s) * HD_) + hd] = (bf16_t)v;
                } else {
                    out_f32[(size_t)row * D_ + e] = v;
                }
            }
        }
    }
}

// ---------------- RoPE (matches reference exactly: "c"=sin, "s"=cos) ----------------

__global__ __launch_bounds__(256) void rope_kernel(bf16_t* __restrict__ Q,
                                                   bf16_t* __restrict__ K) {
    const int idx = blockIdx.x * 256 + threadIdx.x;  // B*H*S*32 threads
    const int j  = idx & 31;
    const int s  = (idx >> 5) & (S_ - 1);
    const int bh = idx >> 16;
    const size_t base = ((size_t)bh * S_ + s) * HD_;
    const float inv_freq = __expf(-((float)(2 * j) / 64.0f) * 9.2103403719761836f);
    const float ang = (float)s * inv_freq;
    const float c  = __sinf(ang);   // reference's "c" is the sin half of emb
    const float sn = __cosf(ang);   // reference's "s" is the cos half of emb
    {
        float x1 = (float)Q[base + j], x2 = (float)Q[base + j + 32];
        Q[base + j]      = (bf16_t)(x1 * c - x2 * sn);
        Q[base + j + 32] = (bf16_t)(x2 * c + x1 * sn);
    }
    {
        float x1 = (float)K[base + j], x2 = (float)K[base + j + 32];
        K[base + j]      = (bf16_t)(x1 * c - x2 * sn);
        K[base + j + 32] = (bf16_t)(x2 * c + x1 * sn);
    }
}

// ---------------- Flash attention: one WG per (b,h, 128 q-rows) ----------------
// 8 waves x 16 q-rows. Keys in blocks of 64, double-buffered:
//   K block (64x64) -> LDS via 2x global_load_async_to_lds_b128 per thread
//   V block         -> LDS transposed via early 32B global load + ds stores
// One s_wait_asynccnt + one barrier per 64-key block (32 total).

__global__ __launch_bounds__(256) void attn_kernel(const bf16_t* __restrict__ Q,
                                                   const bf16_t* __restrict__ K,
                                                   const bf16_t* __restrict__ V,
                                                   bf16_t* __restrict__ ctx) {
    __shared__ bf16_t Ks[2][64][64];    // K block, row-major (row stride 128B)
    __shared__ bf16_t VT[2][64][72];    // V^T block [d][key] (row stride 144B)
    __shared__ bf16_t Ps[8][16][72];    // per-wave P tile (16x64)

    const int lane = threadIdx.x & 31;
    const int wave = threadIdx.x >> 5;
    const int bh = blockIdx.y;
    const int b = bh >> 4, h = bh & 15;
    const size_t base = (size_t)bh * S_ * HD_;
    const bf16_t* Qb = Q + base;
    const bf16_t* Kb = K + base;
    const bf16_t* Vb = V + base;
    const int q0 = blockIdx.x * 128 + wave * 16;

    // staging coordinates: 256 threads cover 64 keys x 64 dims in 32B pieces
    const int skey = threadIdx.x >> 2;        // 0..63
    const int sd0  = (threadIdx.x & 3) * 16;  // 0,16,32,48

    // Q fragments (16x64 as two 16x32), pre-scaled by 1/sqrt(64)=0.125 (exact in bf16)
    v16bf qa0 = load_a_frag(Qb, HD_, q0, 0, lane);
    v16bf qa1 = load_a_frag(Qb, HD_, q0, 32, lane);
#pragma unroll
    for (int i = 0; i < 16; ++i) {
        qa0[i] = (bf16_t)((float)qa0[i] * 0.125f);
        qa1[i] = (bf16_t)((float)qa1[i] * 0.125f);
    }

    v8f c0 = (v8f)(0.0f), c1 = (v8f)(0.0f), c2 = (v8f)(0.0f), c3 = (v8f)(0.0f);
    float mrow[8], lrow[8];
#pragma unroll
    for (int r = 0; r < 8; ++r) { mrow[r] = -__builtin_inff(); lrow[r] = 0.0f; }

    const int rhalf = lane >> 4, cl = lane & 15;

    // prologue: stage block 0 into buffer 0
    {
        const size_t g = (size_t)skey * HD_ + sd0;
        async_copy_b128(Kb + g,     &Ks[0][skey][sd0]);
        async_copy_b128(Kb + g + 8, &Ks[0][skey][sd0 + 8]);
        v16bf vv = *(const v16bf*)(Vb + g);
#pragma unroll
        for (int i = 0; i < 16; ++i) VT[0][sd0 + i][skey] = vv[i];
    }
    wait_asynccnt0();
    __syncthreads();

    const int NBLK = S_ / 64;
    for (int it = 0; it < NBLK; ++it) {
        const int buf = it & 1;
        const bool has_next = (it + 1 < NBLK);
        v16bf vnext;
        if (has_next) {
            const size_t nb = (size_t)((it + 1) * 64 + skey) * HD_ + sd0;
            vnext = *(const v16bf*)(Vb + nb);                          // early V load
            async_copy_b128(Kb + nb,     &Ks[buf ^ 1][skey][sd0]);     // async K
            async_copy_b128(Kb + nb + 8, &Ks[buf ^ 1][skey][sd0 + 8]);
        }

        // scores tile 16x64 = four 16x16 accumulators, K from LDS
        v8f s0 = (v8f)(0.0f), s1 = (v8f)(0.0f), s2 = (v8f)(0.0f), s3 = (v8f)(0.0f);
        {
            const bf16_t* kp = &Ks[buf][0][0];
            v16bf bk;
            bk = load_b_frag_lds(kp, 64, 0,  0,  lane); s0 = wmma_bf16(qa0, bk, s0);
            bk = load_b_frag_lds(kp, 64, 0,  32, lane); s0 = wmma_bf16(qa1, bk, s0);
            bk = load_b_frag_lds(kp, 64, 16, 0,  lane); s1 = wmma_bf16(qa0, bk, s1);
            bk = load_b_frag_lds(kp, 64, 16, 32, lane); s1 = wmma_bf16(qa1, bk, s1);
            bk = load_b_frag_lds(kp, 64, 32, 0,  lane); s2 = wmma_bf16(qa0, bk, s2);
            bk = load_b_frag_lds(kp, 64, 32, 32, lane); s2 = wmma_bf16(qa1, bk, s2);
            bk = load_b_frag_lds(kp, 64, 48, 0,  lane); s3 = wmma_bf16(qa0, bk, s3);
            bk = load_b_frag_lds(kp, 64, 48, 32, lane); s3 = wmma_bf16(qa1, bk, s3);
        }

        // online softmax over 64 keys; cross-lane stats via ds_swizzle xor
#pragma unroll
        for (int r = 0; r < 8; ++r) {
            const float v0 = s0[r], v1 = s1[r], v2 = s2[r], v3 = s3[r];
            float mx = fmaxf(fmaxf(v0, v1), fmaxf(v2, v3));
            mx = fmaxf(mx, SWZ_XOR_F(mx, 1));
            mx = fmaxf(mx, SWZ_XOR_F(mx, 2));
            mx = fmaxf(mx, SWZ_XOR_F(mx, 4));
            mx = fmaxf(mx, SWZ_XOR_F(mx, 8));
            const float mnew  = fmaxf(mrow[r], mx);
            const float alpha = __expf(mrow[r] - mnew);
            const float p0 = __expf(v0 - mnew);
            const float p1 = __expf(v1 - mnew);
            const float p2 = __expf(v2 - mnew);
            const float p3 = __expf(v3 - mnew);
            float rs = (p0 + p1) + (p2 + p3);
            rs += SWZ_XOR_F(rs, 1);
            rs += SWZ_XOR_F(rs, 2);
            rs += SWZ_XOR_F(rs, 4);
            rs += SWZ_XOR_F(rs, 8);
            lrow[r] = lrow[r] * alpha + rs;
            mrow[r] = mnew;
            c0[r] *= alpha; c1[r] *= alpha; c2[r] *= alpha; c3[r] *= alpha;
            bf16_t* prow = &Ps[wave][rhalf * 8 + r][0];
            prow[cl]      = (bf16_t)p0;
            prow[cl + 16] = (bf16_t)p1;
            prow[cl + 32] = (bf16_t)p2;
            prow[cl + 48] = (bf16_t)p3;
        }

        // P(16x64) @ V(64x64) accumulated into ctx, V^T from LDS
        {
            v16bf pa0 = load_a_frag_lds(&Ps[wave][0][0], 72, 0,  lane);
            v16bf pa1 = load_a_frag_lds(&Ps[wave][0][0], 72, 32, lane);
            const bf16_t* vp = &VT[buf][0][0];
            v16bf vb;
            vb = load_b_frag_lds(vp, 72, 0,  0,  lane); c0 = wmma_bf16(pa0, vb, c0);
            vb = load_b_frag_lds(vp, 72, 0,  32, lane); c0 = wmma_bf16(pa1, vb, c0);
            vb = load_b_frag_lds(vp, 72, 16, 0,  lane); c1 = wmma_bf16(pa0, vb, c1);
            vb = load_b_frag_lds(vp, 72, 16, 32, lane); c1 = wmma_bf16(pa1, vb, c1);
            vb = load_b_frag_lds(vp, 72, 32, 0,  lane); c2 = wmma_bf16(pa0, vb, c2);
            vb = load_b_frag_lds(vp, 72, 32, 32, lane); c2 = wmma_bf16(pa1, vb, c2);
            vb = load_b_frag_lds(vp, 72, 48, 0,  lane); c3 = wmma_bf16(pa0, vb, c3);
            vb = load_b_frag_lds(vp, 72, 48, 32, lane); c3 = wmma_bf16(pa1, vb, c3);
        }

        // store the early-loaded V block (transposed) into the next buffer
        if (has_next) {
#pragma unroll
            for (int i = 0; i < 16; ++i) VT[buf ^ 1][sd0 + i][skey] = vnext[i];
        }

        wait_asynccnt0();   // my async K writes for block it+1 are done
        __syncthreads();    // everyone finished reading buf and writing buf^1
    }

    // normalize and store ctx in (B,S,D) bf16 layout for the output projection
#pragma unroll
    for (int r = 0; r < 8; ++r) {
        const float inv = 1.0f / lrow[r];
        const int s = q0 + rhalf * 8 + r;
        const size_t o = ((size_t)b * S_ + s) * D_ + h * HD_;
        ctx[o + 0  + cl] = (bf16_t)(c0[r] * inv);
        ctx[o + 16 + cl] = (bf16_t)(c1[r] * inv);
        ctx[o + 32 + cl] = (bf16_t)(c2[r] * inv);
        ctx[o + 48 + cl] = (bf16_t)(c3[r] * inv);
    }
}

// ---------------- launch ----------------

extern "C" void kernel_launch(void* const* d_in, const int* in_sizes, int n_in,
                              void* d_out, int out_size, void* d_ws, size_t ws_size,
                              hipStream_t stream) {
    (void)in_sizes; (void)n_in; (void)out_size; (void)ws_size;

    const float* hidden = (const float*)d_in[0];
    const float* Wq = (const float*)d_in[1];
    const float* bq = (const float*)d_in[2];
    const float* Wk = (const float*)d_in[3];
    const float* bk = (const float*)d_in[4];
    const float* Wv = (const float*)d_in[5];
    const float* bv = (const float*)d_in[6];
    const float* Wo = (const float*)d_in[7];
    const float* bo = (const float*)d_in[8];
    float* out = (float*)d_out;

    // workspace layout (bytes): all offsets 256-aligned; total 88 MiB
    char* ws = (char*)d_ws;
    bf16_t* Xb  = (bf16_t*)ws;                          // 16 MiB
    bf16_t* Wqb = (bf16_t*)(ws + (16u << 20));          // 2 MiB each
    bf16_t* Wkb = Wqb + (size_t)D_ * D_;
    bf16_t* Wvb = Wkb + (size_t)D_ * D_;
    bf16_t* Wob = Wvb + (size_t)D_ * D_;
    bf16_t* Qd  = (bf16_t*)(ws + (24u << 20));          // 16 MiB each
    bf16_t* Kd  = Qd + (size_t)BS_ * D_;
    bf16_t* Vd  = Kd + (size_t)BS_ * D_;
    bf16_t* Ctx = Vd + (size_t)BS_ * D_;

    cvt_f32_bf16<<<(BS_ * D_ / 4 + 255) / 256, 256, 0, stream>>>(hidden, Xb, BS_ * D_);
    cvt_f32_bf16<<<(D_ * D_ / 4 + 255) / 256, 256, 0, stream>>>(Wq, Wqb, D_ * D_);
    cvt_f32_bf16<<<(D_ * D_ / 4 + 255) / 256, 256, 0, stream>>>(Wk, Wkb, D_ * D_);
    cvt_f32_bf16<<<(D_ * D_ / 4 + 255) / 256, 256, 0, stream>>>(Wv, Wvb, D_ * D_);
    cvt_f32_bf16<<<(D_ * D_ / 4 + 255) / 256, 256, 0, stream>>>(Wo, Wob, D_ * D_);

    dim3 ggrid(BS_ / 128, D_ / 128);
    gemm_bf16<<<ggrid, 256, 0, stream>>>(Xb, Wqb, bq, Qd, nullptr, 0);
    gemm_bf16<<<ggrid, 256, 0, stream>>>(Xb, Wkb, bk, Kd, nullptr, 0);
    gemm_bf16<<<ggrid, 256, 0, stream>>>(Xb, Wvb, bv, Vd, nullptr, 0);

    rope_kernel<<<(B_ * H_ * S_ * 32) / 256, 256, 0, stream>>>(Qd, Kd);

    dim3 agrid(S_ / 128, B_ * H_);
    attn_kernel<<<agrid, 256, 0, stream>>>(Qd, Kd, Vd, Ctx);

    gemm_bf16<<<ggrid, 256, 0, stream>>>(Ctx, Wob, bo, nullptr, out, 1);
}
